// QuantumCircuitGNN_12197707120787
// MI455X (gfx1250) — matrix operand order repaired and checked by reading, not compiled, not run on hardware
//
#include <hip/hip_runtime.h>

#define HID 64
#define NODE_F 16
#define EDGE_F 8
#define GLOBAL_F 40
#define N_GATE 32
#define N_LAYERS 4
#define N_CLS 9
#define N_NODES 100000
#define N_EDGES 1600000
#define BATCH 256
#define EPS 1e-5f

#define DEV __device__ __forceinline__

typedef __bf16 bf16x16 __attribute__((ext_vector_type(16)));
typedef float f32x8 __attribute__((ext_vector_type(8)));

union FragA {
    bf16x16 v;
    unsigned short u[16];
    uint4 q[2];
};

DEV unsigned short f2bf(float f) {
    union { float f; unsigned int u; } x; x.f = f;
    unsigned int r = x.u + 0x7FFFu + ((x.u >> 16) & 1u);  // round-to-nearest-even
    return (unsigned short)(r >> 16);
}

DEV void cvt4(unsigned short* u, float4 f) {
    u[0] = f2bf(f.x); u[1] = f2bf(f.y); u[2] = f2bf(f.z); u[3] = f2bf(f.w);
}

DEV f32x8 wmma_bf16(const FragA& a, const FragA& b, f32x8 c) {
    return __builtin_amdgcn_wmma_f32_16x16x32_bf16(
        /*neg_a=*/false, a.v, /*neg_b=*/false, b.v,
        /*c_mod=*/(short)0, c, /*reuse_a=*/false, /*reuse_b=*/false);
}

DEV void atomicMaxF(float* addr, float val) {
    unsigned int* ua = (unsigned int*)addr;
    unsigned int old = *ua;
    while (__uint_as_float(old) < val) {
        unsigned int assumed = old;
        old = atomicCAS(ua, assumed, __float_as_uint(val));
        if (old == assumed) break;
    }
}

struct P66 { const float* p[66]; };

// ---------------------------------------------------------------------------
// Weight fragmentization: B-fragment layout for V_WMMA_F32_16X16X32_BF16.
// Fragment element e of lane L: K = kt*32 + (L>=16)*8 + (e&7) + (e>>3)*16, N = nt*16 + (L&15)
// Stored contiguously: wfrag[layer][frag][lane][e]  (32B per lane per frag).
// Per layer 52 frags: msg1 kt0..4 (0..19), msg2 (20..27), upd1 (28..43), upd2 (44..51)
// ---------------------------------------------------------------------------
__global__ void __launch_bounds__(256) prep_frag_kernel(P66 P, unsigned short* __restrict__ wfrag) {
    int tid = blockIdx.x * 256 + threadIdx.x;
    if (tid >= N_LAYERS * 52 * 512) return;
    int layer = tid / (52 * 512);
    int r = tid % (52 * 512);
    int frag = r >> 9;
    int li = r & 511;
    int lane = li >> 4, e = li & 15;
    int base = 14 + 11 * layer;
    const float* W; int din, kt, nt;
    if (frag < 20)      { int f = frag;      kt = f >> 2; nt = f & 3; din = 136; W = P.p[base + 4];  }
    else if (frag < 28) { int f = frag - 20; kt = f >> 2; nt = f & 3; din = 64;  W = P.p[base + 6];  }
    else if (frag < 44) { int f = frag - 28; kt = f >> 2; nt = f & 3; din = 128; W = P.p[base + 8];  }
    else                { int f = frag - 44; kt = f >> 2; nt = f & 3; din = 64;  W = P.p[base + 10]; }
    int K = kt * 32 + (lane >> 4) * 8 + (e & 7) + (e >> 3) * 16;
    int col = nt * 16 + (lane & 15);
    wfrag[tid] = (K < din) ? f2bf(W[K * HID + col]) : (unsigned short)0;
}

__global__ void __launch_bounds__(256) prep_gate_kernel(P66 P, unsigned short* __restrict__ gate_bf) {
    int tid = blockIdx.x * 256 + threadIdx.x;
    if (tid >= N_LAYERS * N_GATE * HID) return;
    int layer = tid >> 11;
    int i = tid & 2047;
    gate_bf[tid] = f2bf(P.p[14 + 11 * layer][i]);
}

// ---------------------------------------------------------------------------
// Node embedding: h = LN(ReLU(x @ W + b)); one wave per node, 2 cols per lane.
// ---------------------------------------------------------------------------
__global__ void __launch_bounds__(256) node_embed_kernel(
    const float* __restrict__ x, const float* __restrict__ W, const float* __restrict__ b,
    const float* __restrict__ g, const float* __restrict__ bln,
    float* __restrict__ h, unsigned short* __restrict__ h_bf) {
    int wave = threadIdx.x >> 5, lane = threadIdx.x & 31;
    int n = blockIdx.x * 8 + wave;
    int c0 = lane, c1 = lane + 32;
    float a0 = b[c0], a1 = b[c1];
    for (int k = 0; k < NODE_F; ++k) {
        float xv = x[n * NODE_F + k];
        a0 += xv * W[k * HID + c0];
        a1 += xv * W[k * HID + c1];
    }
    a0 = fmaxf(a0, 0.f); a1 = fmaxf(a1, 0.f);
    float s = a0 + a1;
    for (int m = 16; m >= 1; m >>= 1) s += __shfl_xor(s, m, 32);
    float mean = s * (1.f / 64.f);
    float d0 = a0 - mean, d1 = a1 - mean;
    float q = d0 * d0 + d1 * d1;
    for (int m = 16; m >= 1; m >>= 1) q += __shfl_xor(q, m, 32);
    float inv = rsqrtf(q * (1.f / 64.f) + EPS);
    float r0 = d0 * inv * g[c0] + bln[c0];
    float r1 = d1 * inv * g[c1] + bln[c1];
    h[n * HID + c0] = r0; h[n * HID + c1] = r1;
    h_bf[n * HID + c0] = f2bf(r0); h_bf[n * HID + c1] = f2bf(r1);
}

__global__ void __launch_bounds__(256) zero_kernel(float4* __restrict__ p, int n4) {
    int i = blockIdx.x * 256 + threadIdx.x;
    if (i < n4) p[i] = make_float4(0.f, 0.f, 0.f, 0.f);
}

// ---------------------------------------------------------------------------
// Edge message kernel: one wave per 16-edge tile.
// msg = msg2(ReLU(msg1([h[src] | gate_embed | edge_attr]))) ; atomic-add into agg[dst].
// ---------------------------------------------------------------------------
__global__ void __launch_bounds__(256) edge_msg_kernel(
    const unsigned short* __restrict__ h_bf,
    const unsigned short* __restrict__ wfrag,   // this layer's 52 fragments
    const unsigned short* __restrict__ gate_bf, // this layer's [32][64] bf16 table
    const int* __restrict__ eidx,
    const float* __restrict__ eattr,
    const int* __restrict__ egt,
    const float* __restrict__ msg1_b,
    const float* __restrict__ msg2_b,
    float* __restrict__ agg) {
    __shared__ uint4 ldsq[8][128];  // 2KB bf16 staging per wave
    const int wave = threadIdx.x >> 5;
    const int lane = threadIdx.x & 31;
    const int row  = lane & 15;
    const int half = lane >> 4;
    const int tile = blockIdx.x * 8 + wave;           // N_EDGES/16 = 100000 tiles
    const int e    = tile * 16 + row;
    const int srcI = eidx[e];
    const int dstI = eidx[N_EDGES + e];
    const int gt   = egt[e];

    f32x8 acc[4] = {};
#pragma unroll
    for (int kt = 0; kt < 5; ++kt) {
        FragA a;
        if (kt < 2) {
            const unsigned short* p = h_bf + (size_t)srcI * HID + kt * 32 + half * 8;
            a.q[0] = *(const uint4*)p;
            a.q[1] = *(const uint4*)(p + 16);
        } else if (kt < 4) {
            const unsigned short* p = gate_bf + gt * HID + (kt - 2) * 32 + half * 8;
            a.q[0] = *(const uint4*)p;
            a.q[1] = *(const uint4*)(p + 16);
        } else {
            if (half == 0) {
                float4 f0 = *(const float4*)(eattr + (size_t)e * EDGE_F);
                float4 f1 = *(const float4*)(eattr + (size_t)e * EDGE_F + 4);
                cvt4(&a.u[0], f0); cvt4(&a.u[4], f1);
                a.q[1] = make_uint4(0u, 0u, 0u, 0u);   // K=144..151 padding
            } else {
                a.q[0] = make_uint4(0u, 0u, 0u, 0u);   // K=136..143, 152..159 padding
                a.q[1] = make_uint4(0u, 0u, 0u, 0u);
            }
        }
#pragma unroll
        for (int nt = 0; nt < 4; ++nt) {
            FragA b;
            const unsigned short* wp = wfrag + ((kt * 4 + nt) * 512 + lane * 16);
            b.q[0] = *(const uint4*)wp;
            b.q[1] = *(const uint4*)(wp + 8);
            acc[nt] = wmma_bf16(a, b, acc[nt]);
        }
    }

    // bias + ReLU, stage hidden (16 x 64 bf16) in LDS for msg2 A-fragments
    unsigned short* L = (unsigned short*)ldsq[wave];
#pragma unroll
    for (int nt = 0; nt < 4; ++nt) {
        const float bias = msg1_b[nt * 16 + row];
#pragma unroll
        for (int r = 0; r < 8; ++r) {
            float v = acc[nt][r] + bias;
            v = v > 0.f ? v : 0.f;
            L[(r + half * 8) * HID + nt * 16 + row] = f2bf(v);
        }
    }
    // same-wave producer/consumer: DS pipeline is in-order, compiler inserts waits

    f32x8 dacc[4] = {};
#pragma unroll
    for (int kt = 0; kt < 2; ++kt) {
        FragA a;
        const unsigned short* lp = L + row * HID + kt * 32 + half * 8;
        a.q[0] = *(const uint4*)lp;
        a.q[1] = *(const uint4*)(lp + 16);
#pragma unroll
        for (int nt = 0; nt < 4; ++nt) {
            FragA b;
            const unsigned short* wp = wfrag + ((20 + kt * 4 + nt) * 512 + lane * 16);
            b.q[0] = *(const uint4*)wp;
            b.q[1] = *(const uint4*)(wp + 8);
            dacc[nt] = wmma_bf16(a, b, dacc[nt]);
        }
    }

#pragma unroll
    for (int nt = 0; nt < 4; ++nt) {
        const int col = nt * 16 + row;
        const float bias = msg2_b[col];
#pragma unroll
        for (int r = 0; r < 8; ++r) {
            const int rr = r + half * 8;
            const int dn = __shfl(dstI, rr, 32);
            atomicAdd(agg + (size_t)dn * HID + col, dacc[nt][r] + bias);
        }
    }
}

// ---------------------------------------------------------------------------
// Node update kernel: out = upd2(ReLU(upd1([h | agg]))) ; one wave per 16-node tile.
// ---------------------------------------------------------------------------
__global__ void __launch_bounds__(256) node_upd_kernel(
    const unsigned short* __restrict__ h_bf,
    const float* __restrict__ agg,
    const unsigned short* __restrict__ wfrag,
    const float* __restrict__ upd1_b,
    const float* __restrict__ upd2_b,
    float* __restrict__ outb) {
    __shared__ uint4 ldsq[8][128];
    const int wave = threadIdx.x >> 5;
    const int lane = threadIdx.x & 31;
    const int row  = lane & 15;
    const int half = lane >> 4;
    const int tile = blockIdx.x * 8 + wave;
    if (tile >= N_NODES / 16) return;   // wave-uniform guard
    const int n = tile * 16 + row;

    f32x8 acc[4] = {};
#pragma unroll
    for (int kt = 0; kt < 4; ++kt) {
        FragA a;
        if (kt < 2) {
            const unsigned short* p = h_bf + (size_t)n * HID + kt * 32 + half * 8;
            a.q[0] = *(const uint4*)p;
            a.q[1] = *(const uint4*)(p + 16);
        } else {
            const float* ag = agg + (size_t)n * HID + (kt - 2) * 32 + half * 8;
            float4 f0 = *(const float4*)(ag + 0);
            float4 f1 = *(const float4*)(ag + 4);
            float4 f2 = *(const float4*)(ag + 16);
            float4 f3 = *(const float4*)(ag + 20);
            cvt4(&a.u[0], f0); cvt4(&a.u[4], f1);
            cvt4(&a.u[8], f2); cvt4(&a.u[12], f3);
        }
#pragma unroll
        for (int nt = 0; nt < 4; ++nt) {
            FragA b;
            const unsigned short* wp = wfrag + ((28 + kt * 4 + nt) * 512 + lane * 16);
            b.q[0] = *(const uint4*)wp;
            b.q[1] = *(const uint4*)(wp + 8);
            acc[nt] = wmma_bf16(a, b, acc[nt]);
        }
    }

    unsigned short* L = (unsigned short*)ldsq[wave];
#pragma unroll
    for (int nt = 0; nt < 4; ++nt) {
        const float bias = upd1_b[nt * 16 + row];
#pragma unroll
        for (int r = 0; r < 8; ++r) {
            float v = acc[nt][r] + bias;
            v = v > 0.f ? v : 0.f;
            L[(r + half * 8) * HID + nt * 16 + row] = f2bf(v);
        }
    }

    f32x8 dacc[4] = {};
#pragma unroll
    for (int kt = 0; kt < 2; ++kt) {
        FragA a;
        const unsigned short* lp = L + row * HID + kt * 32 + half * 8;
        a.q[0] = *(const uint4*)lp;
        a.q[1] = *(const uint4*)(lp + 16);
#pragma unroll
        for (int nt = 0; nt < 4; ++nt) {
            FragA b;
            const unsigned short* wp = wfrag + ((44 + kt * 4 + nt) * 512 + lane * 16);
            b.q[0] = *(const uint4*)wp;
            b.q[1] = *(const uint4*)(wp + 8);
            dacc[nt] = wmma_bf16(a, b, dacc[nt]);
        }
    }

#pragma unroll
    for (int nt = 0; nt < 4; ++nt) {
        const int col = nt * 16 + row;
        const float bias = upd2_b[col];
#pragma unroll
        for (int r = 0; r < 8; ++r) {
            const int rr = r + half * 8;
            outb[(size_t)(tile * 16 + rr) * HID + col] = dacc[nt][r] + bias;
        }
    }
}

// ---------------------------------------------------------------------------
// h = h + LayerNorm(out);  one wave per node.
// ---------------------------------------------------------------------------
__global__ void __launch_bounds__(256) ln_residual_kernel(
    const float* __restrict__ outb, const float* __restrict__ g, const float* __restrict__ bln,
    float* __restrict__ h, unsigned short* __restrict__ h_bf) {
    int wave = threadIdx.x >> 5, lane = threadIdx.x & 31;
    int n = blockIdx.x * 8 + wave;
    float o0 = outb[n * HID + lane];
    float o1 = outb[n * HID + lane + 32];
    float s = o0 + o1;
    for (int m = 16; m >= 1; m >>= 1) s += __shfl_xor(s, m, 32);
    float mean = s * (1.f / 64.f);
    float d0 = o0 - mean, d1 = o1 - mean;
    float q = d0 * d0 + d1 * d1;
    for (int m = 16; m >= 1; m >>= 1) q += __shfl_xor(q, m, 32);
    float inv = rsqrtf(q * (1.f / 64.f) + EPS);
    float r0 = h[n * HID + lane]      + d0 * inv * g[lane]      + bln[lane];
    float r1 = h[n * HID + lane + 32] + d1 * inv * g[lane + 32] + bln[lane + 32];
    h[n * HID + lane] = r0;      h[n * HID + lane + 32] = r1;
    h_bf[n * HID + lane] = f2bf(r0); h_bf[n * HID + lane + 32] = f2bf(r1);
}

// ---------------------------------------------------------------------------
// Readout
// ---------------------------------------------------------------------------
__global__ void __launch_bounds__(256) init_readout_kernel(
    float* __restrict__ h_sum, float* __restrict__ h_max, float* __restrict__ cnt) {
    int i = blockIdx.x * 256 + threadIdx.x;
    if (i < BATCH * HID) { h_sum[i] = 0.f; h_max[i] = -__builtin_huge_valf(); }
    if (i < BATCH) cnt[i] = 0.f;
}

__global__ void __launch_bounds__(256) scatter_kernel(
    const float* __restrict__ h, const int* __restrict__ batch,
    float* __restrict__ h_sum, float* __restrict__ h_max, float* __restrict__ cnt) {
    int tid = blockIdx.x * 256 + threadIdx.x;
    if (tid >= N_NODES * HID) return;
    int n = tid >> 6, c = tid & 63;
    int b = batch[n];
    float v = h[tid];
    atomicAdd(h_sum + b * HID + c, v);
    atomicMaxF(h_max + b * HID + c, v);
    if (c == 0) atomicAdd(cnt + b, 1.0f);
}

__global__ void __launch_bounds__(256) gproj_kernel(
    const float* __restrict__ gf, const float* __restrict__ W, const float* __restrict__ b,
    const float* __restrict__ g, const float* __restrict__ bln, float* __restrict__ gout) {
    int wave = threadIdx.x >> 5, lane = threadIdx.x & 31;
    int n = blockIdx.x * 8 + wave;
    int c0 = lane, c1 = lane + 32;
    float a0 = b[c0], a1 = b[c1];
    for (int k = 0; k < GLOBAL_F; ++k) {
        float xv = gf[n * GLOBAL_F + k];
        a0 += xv * W[k * HID + c0];
        a1 += xv * W[k * HID + c1];
    }
    a0 = fmaxf(a0, 0.f); a1 = fmaxf(a1, 0.f);
    float s = a0 + a1;
    for (int m = 16; m >= 1; m >>= 1) s += __shfl_xor(s, m, 32);
    float mean = s * (1.f / 64.f);
    float d0 = a0 - mean, d1 = a1 - mean;
    float q = d0 * d0 + d1 * d1;
    for (int m = 16; m >= 1; m >>= 1) q += __shfl_xor(q, m, 32);
    float inv = rsqrtf(q * (1.f / 64.f) + EPS);
    gout[n * HID + c0] = d0 * inv * g[c0] + bln[c0];
    gout[n * HID + c1] = d1 * inv * g[c1] + bln[c1];
}

__global__ void __launch_bounds__(128) final_kernel(
    const float* __restrict__ h_sum, const float* __restrict__ h_max,
    const float* __restrict__ cnt, const float* __restrict__ g,
    const float* __restrict__ c1b, const float* __restrict__ c1w,
    const float* __restrict__ c2b, const float* __restrict__ c2w,
    const float* __restrict__ thrb, const float* __restrict__ thrw,
    const float* __restrict__ rtb, const float* __restrict__ rtw,
    float* __restrict__ out) {
    __shared__ float comb[256], hid1[128], hid2[64];
    int gid = blockIdx.x, t = threadIdx.x;
    for (int i = t; i < 256; i += 128) {
        int seg = i >> 6, c = i & 63;
        float v;
        if (seg == 0)      { float ct = cnt[gid]; ct = ct > 1.f ? ct : 1.f; v = h_sum[gid * HID + c] / ct; }
        else if (seg == 1) v = h_max[gid * HID + c];
        else if (seg == 2) v = h_sum[gid * HID + c];
        else               v = g[gid * HID + c];
        comb[i] = v;
    }
    __syncthreads();
    {
        float a = c1b[t];
        for (int k = 0; k < 256; ++k) a += comb[k] * c1w[k * 128 + t];
        hid1[t] = a > 0.f ? a : 0.f;
    }
    __syncthreads();
    if (t < 64) {
        float a = c2b[t];
        for (int k = 0; k < 128; ++k) a += hid1[k] * c2w[k * 64 + t];
        hid2[t] = a > 0.f ? a : 0.f;
    }
    __syncthreads();
    if (t < N_CLS) {
        float a = thrb[t];
        for (int k = 0; k < 64; ++k) a += hid2[k] * thrw[k * N_CLS + t];
        out[gid * N_CLS + t] = a;
    }
    if (t == N_CLS) {
        float a = rtb[0];
        for (int k = 0; k < 64; ++k) a += hid2[k] * rtw[k];
        out[BATCH * N_CLS + gid] = a;
    }
}

// ---------------------------------------------------------------------------
// Launch.  d_in order (jax pytree, sorted dict keys):
// 0 x, 1 edge_index, 2 edge_attr, 3 edge_gate_type, 4 batch, 5 global_features,
// 6 comb1.b, 7 comb1.w, 8 comb2.b, 9 comb2.w,
// 10 gp.lin.b, 11 gp.lin.w, 12 gp.ln.b, 13 gp.ln.g,
// 14+11l: gate_embed, ln.b, ln.g, msg1.b, msg1.w, msg2.b, msg2.w, upd1.b, upd1.w, upd2.b, upd2.w,
// 58 ne.lin.b, 59 ne.lin.w, 60 ne.ln.b, 61 ne.ln.g, 62 rt.b, 63 rt.w, 64 thr.b, 65 thr.w
// ---------------------------------------------------------------------------
extern "C" void kernel_launch(void* const* d_in, const int* in_sizes, int n_in,
                              void* d_out, int out_size, void* d_ws, size_t ws_size,
                              hipStream_t stream) {
    (void)in_sizes; (void)out_size; (void)ws_size;
    P66 P;
    for (int i = 0; i < 66 && i < n_in; ++i) P.p[i] = (const float*)d_in[i];

    char* ws = (char*)d_ws;
    float*          h       = (float*)(ws + 0);                 // 25.6 MB
    float*          agg     = (float*)(ws + 25600000);          // 25.6 MB
    float*          outb    = (float*)(ws + 51200000);          // 25.6 MB
    unsigned short* h_bf    = (unsigned short*)(ws + 76800000); // 12.8 MB
    unsigned short* wfrag   = (unsigned short*)(ws + 89600000); // 212,992 B
    unsigned short* gate_bf = (unsigned short*)(ws + 89812992); // 16,384 B
    float*          h_sum   = (float*)(ws + 89829376);
    float*          h_max   = h_sum + BATCH * HID;
    float*          cnt     = h_max + BATCH * HID;
    float*          gbuf    = cnt + BATCH;

    const int*   eidx  = (const int*)d_in[1];
    const float* eattr = (const float*)d_in[2];
    const int*   egt   = (const int*)d_in[3];
    const int*   batch = (const int*)d_in[4];

    prep_frag_kernel<<<(N_LAYERS * 52 * 512 + 255) / 256, 256, 0, stream>>>(P, wfrag);
    prep_gate_kernel<<<(N_LAYERS * N_GATE * HID + 255) / 256, 256, 0, stream>>>(P, gate_bf);
    node_embed_kernel<<<N_NODES / 8, 256, 0, stream>>>(
        (const float*)d_in[0], P.p[59], P.p[58], P.p[61], P.p[60], h, h_bf);

    for (int l = 0; l < N_LAYERS; ++l) {
        int base = 14 + 11 * l;
        zero_kernel<<<(N_NODES * HID / 4 + 255) / 256, 256, 0, stream>>>((float4*)agg, N_NODES * HID / 4);
        edge_msg_kernel<<<N_EDGES / 16 / 8, 256, 0, stream>>>(
            h_bf, wfrag + l * 26624, gate_bf + l * 2048,
            eidx, eattr, egt, P.p[base + 3], P.p[base + 5], agg);
        node_upd_kernel<<<(N_NODES / 16 + 7) / 8, 256, 0, stream>>>(
            h_bf, agg, wfrag + l * 26624, P.p[base + 7], P.p[base + 9], outb);
        ln_residual_kernel<<<N_NODES / 8, 256, 0, stream>>>(
            outb, P.p[base + 2], P.p[base + 1], h, h_bf);
    }

    init_readout_kernel<<<(BATCH * HID + 255) / 256, 256, 0, stream>>>(h_sum, h_max, cnt);
    scatter_kernel<<<(N_NODES * HID + 255) / 256, 256, 0, stream>>>(h, batch, h_sum, h_max, cnt);
    gproj_kernel<<<BATCH / 8, 256, 0, stream>>>(P.p[5], P.p[11], P.p[10], P.p[13], P.p[12], gbuf);
    final_kernel<<<BATCH, 128, 0, stream>>>(
        h_sum, h_max, cnt, gbuf,
        P.p[6], P.p[7], P.p[8], P.p[9], P.p[64], P.p[65], P.p[62], P.p[63],
        (float*)d_out);
}
